// LocalAtomAttention_44487271252527
// MI455X (gfx1250) — matrix operand order repaired and checked by reading, not compiled, not run on hardware
//
#include <hip/hip_runtime.h>

// ---------------------------------------------------------------------------
// LocalAtomAttention for MI455X (gfx1250, wave32, WMMA bf16)
//
// Shapes: x[B=16][L=4096][A=4][C=128] f32, mask[B][L][A] int, W* 128x128 f32.
// One 16-row tile = 4 residues x 4 atoms. All GEMMs via v_wmma_f32_16x16x32_bf16.
// f32->bf16 via native __bf16 casts (v_cvt_pk_bf16_f32 on gfx1250).
// ---------------------------------------------------------------------------

typedef __attribute__((ext_vector_type(16))) __bf16       v16bf;
typedef __attribute__((ext_vector_type(8)))  float        v8f;
typedef __attribute__((ext_vector_type(4)))  float        v4f;
typedef __attribute__((ext_vector_type(4)))  unsigned int v4u;
typedef __attribute__((ext_vector_type(2)))  unsigned int v2u;
typedef __attribute__((ext_vector_type(4)))  int          v4i;

#define B_DIM 16
#define L_DIM 4096
#define A_DIM 4
#define C_DIM 128
#define SCALE 0.17677669529663687f /* 1/sqrt(32) */

#define WAVES_PER_BLOCK 8
#define TILES_PER_WAVE  8
#define THREADS   (WAVES_PER_BLOCK * 32)
#define ROWS_TOTAL (B_DIM * L_DIM * A_DIM)                 /* 262144 */
#define TILES_TOTAL (ROWS_TOTAL / 16)                      /* 16384  */
#define NBLOCKS (TILES_TOTAL / (WAVES_PER_BLOCK * TILES_PER_WAVE)) /* 256 */

// ---- LDS layout (bytes) ----
#define LDS_W(i)  ((i) * 32768)   /* Wq,Wk,Wv,Wo bf16 row-major, 128x128 each */
#define LDS_BIAS  131072          /* bq|bk|bv|bo f32, 4*128 */
#define LDS_WAVE  133120
#define WSCR      16896
#define OFF_Q     0               /* 16x128 bf16, row-major      (4096 B) */
#define OFF_K     4096            /* 16x128 bf16, row-major      (4096 B) */
#define OFF_VT    8192            /* 128x16 bf16, V transposed   (4096 B) */
#define OFF_ATT   12288           /* 16x16  bf16                 ( 512 B) */
#define OFF_AO    12800           /* 16x128 bf16, attn output    (4096 B) */
#define LDS_TOTAL (LDS_WAVE + WAVES_PER_BLOCK * WSCR)      /* 268288 */

union BF16Frag {
  v16bf v;
  v4u   q[2];
};
union BF16x8 {
  __bf16 h[8];
  v4u    q;
};
union BF16x4 {
  __bf16 h[4];
  v2u    q;
};

extern "C" __global__ void __launch_bounds__(THREADS, 1)
LocalAtomAttention_44487271252527_kernel(
    const float* __restrict__ x, const int* __restrict__ mask,
    const float* __restrict__ Wq, const float* __restrict__ bq,
    const float* __restrict__ Wk, const float* __restrict__ bk,
    const float* __restrict__ Wv, const float* __restrict__ bv,
    const float* __restrict__ Wo, const float* __restrict__ bo,
    float* __restrict__ out) {
  extern __shared__ unsigned char smem[];
  const int tid = threadIdx.x;

  // ---------------- stage weights (f32 -> bf16) and biases into LDS --------
  {
    const float* Wsrc[4] = {Wq, Wk, Wv, Wo};
    for (int w = 0; w < 4; ++w) {
      const v4f* src = (const v4f*)Wsrc[w];
      v2u* dst = (v2u*)(smem + LDS_W(w));
      for (int i = tid; i < C_DIM * C_DIM / 4; i += THREADS) {
        v4f f = src[i];
        BF16x4 p;
        p.h[0] = (__bf16)f.x; p.h[1] = (__bf16)f.y;
        p.h[2] = (__bf16)f.z; p.h[3] = (__bf16)f.w;
        dst[i] = p.q;
      }
    }
    float* bs = (float*)(smem + LDS_BIAS);
    for (int i = tid; i < 4 * C_DIM; i += THREADS) {
      float v = (i < 128) ? bq[i] : (i < 256) ? bk[i - 128]
              : (i < 384) ? bv[i - 256] : bo[i - 384];
      bs[i] = v;
    }
  }
  __syncthreads();

  const int wid  = tid >> 5;
  const int lane = tid & 31;
  const int m    = lane & 15;   // row (A) / col (B,C,D) index within tile
  const int kh   = lane >> 4;   // lane-half selector

  __bf16* Qs   = (__bf16*)(smem + LDS_WAVE + wid * WSCR + OFF_Q);
  __bf16* Ks   = (__bf16*)(smem + LDS_WAVE + wid * WSCR + OFF_K);
  __bf16* VTs  = (__bf16*)(smem + LDS_WAVE + wid * WSCR + OFF_VT);
  __bf16* ATTs = (__bf16*)(smem + LDS_WAVE + wid * WSCR + OFF_ATT);
  __bf16* AOs  = (__bf16*)(smem + LDS_WAVE + wid * WSCR + OFF_AO);
  const float* bias = (const float*)(smem + LDS_BIAS);

  for (int ti = 0; ti < TILES_PER_WAVE; ++ti) {
    const int tile = (blockIdx.x * WAVES_PER_BLOCK + wid) * TILES_PER_WAVE + ti;
    const int row0 = tile * 16;                         // global atom-row base
    const int b    = row0 >> 14;                        // / (L*A)
    const int l0   = (row0 & (L_DIM * A_DIM - 1)) >> 2; // first residue of tile

    if (ti + 1 < TILES_PER_WAVE)
      __builtin_prefetch(x + (size_t)(row0 + 16 + m) * C_DIM, 0, 0);

    // ---------------- x A-fragments (global f32 -> bf16, A layout) --------
    const float* xrow = x + (size_t)(row0 + m) * C_DIM;
    BF16Frag a[4];
#pragma unroll
    for (int c = 0; c < 4; ++c) {
      const float* p0 = xrow + c * 32 + 8 * kh;        // K = 8*kh ..
      const float* p1 = xrow + c * 32 + 16 + 8 * kh;   // K = 16+8*kh ..
      v4f f0 = *(const v4f*)p0, f1 = *(const v4f*)(p0 + 4);
      v4f f2 = *(const v4f*)p1, f3 = *(const v4f*)(p1 + 4);
      v16bf av;
      av[0]  = (__bf16)f0.x; av[1]  = (__bf16)f0.y;
      av[2]  = (__bf16)f0.z; av[3]  = (__bf16)f0.w;
      av[4]  = (__bf16)f1.x; av[5]  = (__bf16)f1.y;
      av[6]  = (__bf16)f1.z; av[7]  = (__bf16)f1.w;
      av[8]  = (__bf16)f2.x; av[9]  = (__bf16)f2.y;
      av[10] = (__bf16)f2.z; av[11] = (__bf16)f2.w;
      av[12] = (__bf16)f3.x; av[13] = (__bf16)f3.y;
      av[14] = (__bf16)f3.z; av[15] = (__bf16)f3.w;
      a[c].v = av;
    }

    // ---------------- masks ----------------------------------------------
    const v4i* mp = (const v4i*)mask;                  // one v4i per residue
    v4i mres[4];
#pragma unroll
    for (int r = 0; r < 4; ++r) mres[r] = mp[(size_t)b * L_DIM + l0 + r];
    const int col_res  = m >> 2;
    const int col_atom = m & 3;
    const v4i mc = mres[col_res];
    const int kvalid = (col_atom == 0) ? mc.x : (col_atom == 1) ? mc.y
                     : (col_atom == 2) ? mc.z : mc.w;

    // ---------------- Q/K/V projections (WMMA over K=128) -----------------
#pragma unroll
    for (int p = 0; p < 3; ++p) {
      const __bf16* Wb = (const __bf16*)(smem + LDS_W(p));
      const float* bb = bias + p * C_DIM;
      for (int nt = 0; nt < 8; ++nt) {
        const int ch = nt * 16 + m;                    // output channel
        const float bv0 = bb[ch];
        v8f acc;
#pragma unroll
        for (int i = 0; i < 8; ++i) acc[i] = bv0;
#pragma unroll
        for (int c = 0; c < 4; ++c) {
          BF16Frag bf;
          const __bf16* wp = Wb + ch * C_DIM + c * 32 + 8 * kh;
          bf.q[0] = *(const v4u*)wp;
          bf.q[1] = *(const v4u*)(wp + 16);
          acc = __builtin_amdgcn_wmma_f32_16x16x32_bf16(
              false, a[c].v, false, bf.v, (short)0, acc, false, false);
        }
        if (p == 0) {                                  // Q: row-major
#pragma unroll
          for (int v = 0; v < 8; ++v)
            Qs[(v + 8 * kh) * C_DIM + ch] = (__bf16)acc[v];
        } else if (p == 1) {                           // K: row-major
#pragma unroll
          for (int v = 0; v < 8; ++v)
            Ks[(v + 8 * kh) * C_DIM + ch] = (__bf16)acc[v];
        } else {                                       // V: transposed [ch][16]
          BF16x8 pv;
#pragma unroll
          for (int v = 0; v < 8; ++v) pv.h[v] = (__bf16)acc[v];
          *(v4u*)(VTs + ch * 16 + 8 * kh) = pv.q;
        }
      }
    }
    asm volatile("" ::: "memory");  // DS in-order per wave; block reordering

    // ---------------- per-head attention ----------------------------------
#pragma unroll
    for (int h = 0; h < 4; ++h) {
      BF16Frag qa;                                     // A = Q_h (16x32)
      const __bf16* qp = Qs + m * C_DIM + 32 * h + 8 * kh;
      qa.q[0] = *(const v4u*)qp;
      qa.q[1] = *(const v4u*)(qp + 16);
      BF16Frag kb;                                     // B = K_h^T (32x16)
      const __bf16* kp = Ks + m * C_DIM + 32 * h + 8 * kh;
      kb.q[0] = *(const v4u*)kp;
      kb.q[1] = *(const v4u*)(kp + 16);
      v8f sc;
#pragma unroll
      for (int i = 0; i < 8; ++i) sc[i] = 0.0f;
      sc = __builtin_amdgcn_wmma_f32_16x16x32_bf16(
          false, qa.v, false, kb.v, (short)0, sc, false, false);

      // block-diagonal mask + softmax over the 16-lane row (per 4-residue tile,
      // off-residue columns are -inf so this equals the per-residue softmax)
      float e[8];
#pragma unroll
      for (int v = 0; v < 8; ++v) {
        const int row_res = (v + 8 * kh) >> 2;
        float s = (row_res == col_res && kvalid) ? sc[v] * SCALE : -1e30f;
        float mx = s;
        mx = fmaxf(mx, __shfl_xor(mx, 1));
        mx = fmaxf(mx, __shfl_xor(mx, 2));
        mx = fmaxf(mx, __shfl_xor(mx, 4));
        mx = fmaxf(mx, __shfl_xor(mx, 8));
        float ev = __expf(s - mx);
        float sum = ev;
        sum += __shfl_xor(sum, 1);
        sum += __shfl_xor(sum, 2);
        sum += __shfl_xor(sum, 4);
        sum += __shfl_xor(sum, 8);
        e[v] = ev / sum;
      }
#pragma unroll
      for (int v = 0; v < 8; ++v)
        ATTs[(v + 8 * kh) * 16 + m] = (__bf16)e[v];
      asm volatile("" ::: "memory");

      BF16Frag aa;                                     // A = attn (K padded 16->32)
      aa.q[0] = *(const v4u*)(ATTs + m * 16 + 8 * kh);
      aa.q[1] = (v4u){0, 0, 0, 0};
#pragma unroll
      for (int t = 0; t < 2; ++t) {                    // B = V_h (rows 16..31 zero)
        BF16Frag vb;
        const __bf16* vp = VTs + (32 * h + 16 * t + m) * 16;
        v4u z = {0, 0, 0, 0};
        v4u v0 = *(const v4u*)vp;
        v4u v1 = *(const v4u*)(vp + 8);
        vb.q[0] = kh ? v1 : v0;
        vb.q[1] = z;
        v8f ao;
#pragma unroll
        for (int i = 0; i < 8; ++i) ao[i] = 0.0f;
        ao = __builtin_amdgcn_wmma_f32_16x16x32_bf16(
            false, aa.v, false, vb.v, (short)0, ao, false, false);
        const int ch = (2 * h + t) * 16 + m;
#pragma unroll
        for (int v = 0; v < 8; ++v)
          AOs[(v + 8 * kh) * C_DIM + ch] = (__bf16)ao[v];
      }
    }
    asm volatile("" ::: "memory");

    // ---------------- output projection + residue mask + store -------------
    float rm[4];
#pragma unroll
    for (int r = 0; r < 4; ++r)
      rm[r] = (mres[r].x | mres[r].y | mres[r].z | mres[r].w) ? 1.0f : 0.0f;

    const __bf16* WOb = (const __bf16*)(smem + LDS_W(3));
    const float* bob = bias + 3 * C_DIM;
    for (int nt = 0; nt < 8; ++nt) {
      const int ch = nt * 16 + m;
      const float bv0 = bob[ch];
      v8f acc;
#pragma unroll
      for (int i = 0; i < 8; ++i) acc[i] = bv0;
#pragma unroll
      for (int c = 0; c < 4; ++c) {
        BF16Frag af;
        const __bf16* ap = AOs + m * C_DIM + c * 32 + 8 * kh;
        af.q[0] = *(const v4u*)ap;
        af.q[1] = *(const v4u*)(ap + 16);
        BF16Frag bf;
        const __bf16* wp = WOb + ch * C_DIM + c * 32 + 8 * kh;
        bf.q[0] = *(const v4u*)wp;
        bf.q[1] = *(const v4u*)(wp + 16);
        acc = __builtin_amdgcn_wmma_f32_16x16x32_bf16(
            false, af.v, false, bf.v, (short)0, acc, false, false);
      }
#pragma unroll
      for (int v = 0; v < 8; ++v) {
        const int M = v + 8 * kh;
        out[(size_t)(row0 + M) * C_DIM + ch] = acc[v] * rm[M >> 2];
      }
    }
  }
}

extern "C" void kernel_launch(void* const* d_in, const int* in_sizes, int n_in,
                              void* d_out, int out_size, void* d_ws, size_t ws_size,
                              hipStream_t stream) {
  (void)in_sizes; (void)n_in; (void)out_size; (void)d_ws; (void)ws_size;
  const float* x  = (const float*)d_in[0];
  const int*   mk = (const int*)d_in[1];     // jax bool mask delivered as int32
  const float* Wq = (const float*)d_in[2];
  const float* bq = (const float*)d_in[3];
  const float* Wk = (const float*)d_in[4];
  const float* bk = (const float*)d_in[5];
  const float* Wv = (const float*)d_in[6];
  const float* bv = (const float*)d_in[7];
  const float* Wo = (const float*)d_in[8];
  const float* bo = (const float*)d_in[9];
  float* out = (float*)d_out;

  LocalAtomAttention_44487271252527_kernel
      <<<dim3(NBLOCKS), dim3(THREADS), LDS_TOTAL, stream>>>(
          x, mk, Wq, bq, Wk, bk, Wv, bv, Wo, bo, out);
}
